// MultiViewAggregation_41300405518369
// MI455X (gfx1250) — compile-verified, matrix operand order checked by reading, not run on hardware
//
#include <hip/hip_runtime.h>
#include <hip/hip_bf16.h>

typedef __attribute__((ext_vector_type(16))) _Float16 v16h;
typedef __attribute__((ext_vector_type(8)))  float    v8f;
typedef __attribute__((ext_vector_type(4)))  unsigned int v4u;
typedef __attribute__((ext_vector_type(8)))  int      v8i;
typedef __attribute__((ext_vector_type(4)))  int      v4i;

#define WMMA_F16(A,B,C) __builtin_amdgcn_wmma_f32_16x16x32_f16(false,(A),false,(B),(short)0,(C),false,false)

__device__ __forceinline__ float eluf(float x){ return x>0.f ? x : __expf(x)-1.f; }
__device__ __forceinline__ float geluf(float x){ return 0.5f*x*(1.f+erff(x*0.70710678118654752f)); }

// ---- Pre-converted weight blob offsets (in halfs), laid out contiguously so one
// ---- TDM descriptor can pull a whole per-phase parameter block into LDS.
#define OFF_K1W1 0          // 64x32
#define OFF_K1W2 2048       // 64x64
#define OFF_K1W3 6144       // 64x64
#define OFF_K1W4 10240      // 32x64
#define SZ_K1    12288
#define OFF_TV1  12288      // 32x96
#define OFF_TO1  15360      // 96x96
#define SZ_TV1TO1 12288
#define OFF_N1W1 24576      // 256x96
#define OFF_N1W2 49152      // 96x256
#define OFF_TV2  73728      // 32x96
#define OFF_TO2  76800      // 96x96
#define SZ_TV2TO2 12288
#define OFF_N2W1 86016      // 256x96
#define OFF_N2W2 110592     // 96x256
#define OFF_BW1  135168     // 128x96
#define OFF_BW2  147456     // 128x128
#define WBLOB_HALFS 163840

// ---- Tensor Data Mover: DMA a contiguous f16 block global -> LDS.
// D# per CDNA5 ISA (2D tensor, 8-byte elements, 1 row).
__device__ __forceinline__ void tdm_copy(const void* gsrc, void* lds_dst, unsigned bytes){
  unsigned long long ga = (unsigned long long)gsrc;
  unsigned lds = (unsigned)(unsigned long long)lds_dst;   // flat->LDS: low 32 bits = LDS offset
  unsigned n8 = bytes >> 3;                               // # of 8-byte elements
  v4u g0;
  g0.x = 1u;                                   // count=1, is_restore=0, gather off
  g0.y = lds;                                  // lds_addr [63:32]
  g0.z = (unsigned)ga;                         // global_addr [95:64]
  g0.w = (unsigned)(ga >> 32) | (2u << 30);    // global_addr[56:32] | type=2
  v8i g1;
  g1[0] = (int)(3u << 16);                     // wg_mask=0 | data_size=3 (8B)
  g1[1] = (int)((n8 & 0xFFFFu) << 16);         // tensor_dim0 lo16
  g1[2] = (int)((n8 >> 16) | (1u << 16));      // tensor_dim0 hi16 | tensor_dim1 lo16 = 1
  g1[3] = (int)(n8 << 16);                     // tensor_dim1 hi16 = 0 | tile_dim0 = n8
  g1[4] = 1;                                   // tile_dim1 = 1, tile_dim2 = 0
  g1[5] = (int)n8;                             // tensor_dim0_stride lo32
  g1[6] = 0;
  g1[7] = 0;
  v4i z4 = {0,0,0,0};
#if __clang_major__ >= 23
  v8i z8 = {0,0,0,0,0,0,0,0};
  __builtin_amdgcn_tensor_load_to_lds(g0, g1, z4, z4, z8, 0);
#else
  __builtin_amdgcn_tensor_load_to_lds(g0, g1, z4, z4, 0);
#endif
}

// Issue from one wave, wait TENSORcnt, then make visible to the workgroup.
__device__ __forceinline__ void tdm_stage(const _Float16* g, _Float16* lds, unsigned halfs, int tid){
  if (tid == 0){
    tdm_copy(g, lds, halfs*2u);
    __builtin_amdgcn_s_wait_tensorcnt(0);
  }
  __syncthreads();
}

// CDNA5 16-bit 16x32 A/B fragment load from [row][k] f16 LDS buffer.
// A: row = local M (lane&15). B: row = output column n. half = lane>>4.
// K mapping per ISA: k = kbase + j + (j>=8?8:0) + 8*half ; (k,k+1) pairs contiguous.
__device__ __forceinline__ v16h load_frag(const _Float16* base, int row, int stride,
                                          int kbase, int half) {
  v16h f;
  const _Float16* p = base + row*stride + kbase + half*8;
  #pragma unroll
  for (int pp=0; pp<8; ++pp) {
    int k0 = 2*pp + (pp>=4 ? 8 : 0);
    f[2*pp]   = p[k0];
    f[2*pp+1] = p[k0+1];
  }
  return f;
}

__device__ __forceinline__ void row_stats96(const _Float16* row, int half, float& m, float& rs){
  float s=0.f, s2=0.f;
  #pragma unroll
  for (int c=0;c<48;++c){ float v=(float)row[half*48+c]; s+=v; s2+=v*v; }
  s  += __shfl_xor(s, 16, 32);
  s2 += __shfl_xor(s2, 16, 32);
  m = s*(1.f/96.f);
  float var = s2*(1.f/96.f) - m*m;
  rs = rsqrtf(fmaxf(var,0.f) + 1e-5f);
}

__device__ __forceinline__ v16h ln_frag96(const _Float16* row, int kbase, int half,
                                          float m, float rs,
                                          const float* g, const float* b){
  v16h f;
  #pragma unroll
  for (int j=0;j<16;++j){
    int k = kbase + j + (j>=8?8:0) + half*8;
    f[j] = (_Float16)(((float)row[k]-m)*rs*g[k]+b[k]);
  }
  return f;
}

#define ZERO8 {0.f,0.f,0.f,0.f,0.f,0.f,0.f,0.f}

// ---------------- Kernel 0: one-time weight conversion f32 [K][N] -> f16 [N][Kpad] ----------------
__device__ __forceinline__ void convw(const float* __restrict__ W, _Float16* __restrict__ dst,
                                      int K, int N, int Kpad, int gid, int gsz){
  if (gid*32 < K*N) __builtin_prefetch(W + gid*32, 0, 0);   // global_prefetch_b8
  for (int i=gid; i<N*Kpad; i+=gsz){
    int n=i/Kpad, k=i%Kpad;
    dst[i] = (k<K) ? (_Float16)W[k*N+n] : (_Float16)0.f;
  }
}

__global__ __launch_bounds__(256) void k0_prep(
    const float* W1,const float* W2,const float* W3,const float* W4,
    const float* TV1,const float* TO1,const float* N1W1,const float* N1W2,
    const float* TV2,const float* TO2,const float* N2W1,const float* N2W2,
    const float* BW1,const float* BW2, _Float16* wb)
{
  int gid = blockIdx.x*blockDim.x+threadIdx.x;
  int gsz = gridDim.x*blockDim.x;
  convw(W1,  wb+OFF_K1W1,   7,  64,  32, gid, gsz);
  convw(W2,  wb+OFF_K1W2,  64,  64,  64, gid, gsz);
  convw(W3,  wb+OFF_K1W3,  64,  64,  64, gid, gsz);
  convw(W4,  wb+OFF_K1W4,  64,  32,  64, gid, gsz);
  convw(TV1, wb+OFF_TV1,   96,  32,  96, gid, gsz);
  convw(TO1, wb+OFF_TO1,   96,  96,  96, gid, gsz);
  convw(N1W1,wb+OFF_N1W1,  96, 256,  96, gid, gsz);
  convw(N1W2,wb+OFF_N1W2, 256,  96, 256, gid, gsz);
  convw(TV2, wb+OFF_TV2,   96,  32,  96, gid, gsz);
  convw(TO2, wb+OFF_TO2,   96,  96,  96, gid, gsz);
  convw(N2W1,wb+OFF_N2W1,  96, 256,  96, gid, gsz);
  convw(N2W2,wb+OFF_N2W2, 256,  96, 256, gid, gsz);
  convw(BW1, wb+OFF_BW1,   96, 128,  96, gid, gsz);
  convw(BW2, wb+OFF_BW2,  128, 128, 128, gid, gsz);
}

// ---------------- Kernel 1: PBR MLP, one (pixel,view) tile per wave ----------------
__global__ __launch_bounds__(256) void k1_pbr(
    const float* __restrict__ view_dir, const float* __restrict__ normal,
    const float* __restrict__ DL,
    const float* __restrict__ ln_g, const float* __restrict__ ln_b,
    const float* __restrict__ B1, const float* __restrict__ B2,
    const float* __restrict__ B3, const float* __restrict__ B4,
    const _Float16* __restrict__ wb,
    float* __restrict__ pbrfeat)
{
  __shared__ _Float16 kw[SZ_K1];     // [w1t|w2t|w3t|w4t]
  __shared__ float b1s[64], b2s[64], b3s[64], b4s[32];
  __shared__ _Float16 scr[8][2][16*64];

  const int tid = threadIdx.x, nth = blockDim.x;
  for (int i=tid;i<64;i+=nth){ b1s[i]=B1[i]; b2s[i]=B2[i]; b3s[i]=B3[i]; }
  for (int i=tid;i<32;i+=nth) b4s[i]=B4[i];
  tdm_stage(wb+OFF_K1W1, kw, SZ_K1, tid);     // one TDM for all 4 weight mats (+barrier)

  const _Float16* w1t = kw;
  const _Float16* w2t = kw + (OFF_K1W2-OFF_K1W1);
  const _Float16* w3t = kw + (OFF_K1W3-OFF_K1W1);
  const _Float16* w4t = kw + (OFF_K1W4-OFF_K1W1);

  const int wave=tid>>5, lane=tid&31, half=lane>>4, lrow=lane&15;
  const int pix = blockIdx.x;
  const size_t R = (size_t)pix*8 + wave;

  // Build features + LayerNorm(7), pack into A fragment (K padded 7->32).
  float y[7];
  #pragma unroll
  for (int c=0;c<7;++c) y[c]=0.f;
  if (half==0) {
    float f[7];
    #pragma unroll
    for (int c=0;c<7;++c) f[c]=0.f;
    if (lrow < 12) {
      const float nx=normal[pix*3+0], ny=normal[pix*3+1], nz=normal[pix*3+2];
      const float vx=view_dir[R*3+0], vy=view_dir[R*3+1], vz=view_dir[R*3+2];
      const float* dl = DL + (size_t)pix*84 + lrow*7;
      float dx=dl[0], dy=dl[1], dz=dl[2];
      float hdv = (dx*vx+dy*vy+dz*vz+1.f)*0.5f;
      f[0]=dx*nx+dy*ny+dz*nz;
      f[1]=dl[3]; f[2]=dl[4]; f[3]=dl[5]; f[4]=dl[6];
      f[5]=exp2f((-5.55472f*hdv-6.98316f)*hdv);
      f[6]=vx*nx+vy*ny+vz*nz;
    }
    float m=0.f;
    #pragma unroll
    for (int c=0;c<7;++c) m+=f[c];
    m *= (1.f/7.f);
    float var=0.f;
    #pragma unroll
    for (int c=0;c<7;++c){ float d=f[c]-m; var+=d*d; }
    var *= (1.f/7.f);
    float rs = rsqrtf(var+1e-5f);
    #pragma unroll
    for (int c=0;c<7;++c) y[c]=(f[c]-m)*rs*ln_g[c]+ln_b[c];
  }
  v16h a0;
  #pragma unroll
  for (int j=0;j<16;++j){
    int k = j + (j>=8?8:0) + half*8;
    a0[j] = (half==0 && k<7) ? (_Float16)y[k] : (_Float16)0.f;
  }

  _Float16* sA = scr[wave][0];
  _Float16* sB = scr[wave][1];

  // layer 1: 16x32 @ 32x64
  #pragma unroll
  for (int nt=0; nt<4; ++nt){
    v8f c = ZERO8;
    c = WMMA_F16(a0, load_frag(w1t, nt*16+lrow, 32, 0, half), c);
    float bias = b1s[nt*16+lrow];
    #pragma unroll
    for (int r=0;r<8;++r)
      sA[(r+8*half)*64 + nt*16 + lrow] = (_Float16)eluf(c[r]+bias);
  }
  __syncthreads();

  // layer 2
  {
    v16h k0 = load_frag(sA, lrow, 64, 0, half);
    v16h k1 = load_frag(sA, lrow, 64, 32, half);
    #pragma unroll
    for (int nt=0; nt<4; ++nt){
      v8f c = ZERO8;
      c = WMMA_F16(k0, load_frag(w2t, nt*16+lrow, 64, 0, half), c);
      c = WMMA_F16(k1, load_frag(w2t, nt*16+lrow, 64, 32, half), c);
      float bias = b2s[nt*16+lrow];
      #pragma unroll
      for (int r=0;r<8;++r)
        sB[(r+8*half)*64 + nt*16 + lrow] = (_Float16)eluf(c[r]+bias);
    }
  }
  __syncthreads();

  // layer 3
  {
    v16h k0 = load_frag(sB, lrow, 64, 0, half);
    v16h k1 = load_frag(sB, lrow, 64, 32, half);
    #pragma unroll
    for (int nt=0; nt<4; ++nt){
      v8f c = ZERO8;
      c = WMMA_F16(k0, load_frag(w3t, nt*16+lrow, 64, 0, half), c);
      c = WMMA_F16(k1, load_frag(w3t, nt*16+lrow, 64, 32, half), c);
      float bias = b3s[nt*16+lrow];
      #pragma unroll
      for (int r=0;r<8;++r)
        sA[(r+8*half)*64 + nt*16 + lrow] = (_Float16)eluf(c[r]+bias);
    }
  }
  __syncthreads();

  // layer 4 + sum over 12 SG rows (mask padding rows, combine halves via shfl)
  {
    v16h k0 = load_frag(sA, lrow, 64, 0, half);
    v16h k1 = load_frag(sA, lrow, 64, 32, half);
    #pragma unroll
    for (int nt=0; nt<2; ++nt){
      v8f c = ZERO8;
      c = WMMA_F16(k0, load_frag(w4t, nt*16+lrow, 64, 0, half), c);
      c = WMMA_F16(k1, load_frag(w4t, nt*16+lrow, 64, 32, half), c);
      float s = 0.f;
      const int rmax = half ? 4 : 8;   // rows 12..15 are padding
      #pragma unroll
      for (int r=0;r<8;++r) if (r<rmax) s += c[r];
      s += __shfl_xor(s, 16, 32);
      s += 12.f * b4s[nt*16+lrow];
      if (half==0)
        pbrfeat[R*32 + nt*16 + lrow] = s;
    }
  }
}

// ---------------- Kernel 2: block 1 (tv1 + moments + to1 + MLP 96->256->96) ----------------
__global__ __launch_bounds__(256) void k2_block1(
    const float* __restrict__ rgb, const float* __restrict__ fm,
    const float* __restrict__ proj_err, const float* __restrict__ pbrfeat,
    const float* __restrict__ g1, const float* __restrict__ bL1,
    const float* __restrict__ to1_b,
    const float* __restrict__ g2, const float* __restrict__ bL2,
    const float* __restrict__ n1_b1, const float* __restrict__ n1_b2,
    const _Float16* __restrict__ wb,
    float* __restrict__ xout)
{
  __shared__ _Float16 xin[128*96];
  __shared__ float wgt[128];
  __shared__ float wsum[16];
  __shared__ _Float16 wsh[3072 + 24576];    // [tv1 | big weight buffer]
  __shared__ float bbuf[256];
  __shared__ float lgA[96], lbA[96], lgB[96], lbB[96];
  __shared__ _Float16 wscr[8][16*256];

  _Float16* wv   = wsh;          // 32x96
  _Float16* wbuf = wsh + 3072;   // up to 256x96

  const int tid=threadIdx.x, nth=blockDim.x;
  const size_t Rbase=(size_t)blockIdx.x*128;
  const int pixbase=blockIdx.x*16;

  // x_input = [rgb | featmaps | pbr_feature]
  for (int i=tid; i<128*96; i+=nth){
    int row=i/96, col=i%96;
    size_t R=Rbase+row;
    int pix=pixbase+(row>>3);
    float v;
    if (col<3)        v = rgb[R*3+col];
    else if (col<64)  v = fm[(size_t)pix*61 + (col-3)];
    else              v = pbrfeat[R*32 + (col-64)];
    xin[i]=(_Float16)v;
  }
  if (tid<128){
    float pe = proj_err[Rbase+tid];
    wgt[tid] = -fminf(log10f(fabsf(pe)+1e-6f), 0.f);
  }
  for (int i=tid;i<96;i+=nth){
    bbuf[i]=to1_b[i];
    lgA[i]=g1[i]; lbA[i]=bL1[i];
    lgB[i]=g2[i]; lbB[i]=bL2[i];
  }
  tdm_stage(wb+OFF_TV1, wsh, SZ_TV1TO1, tid);   // [tv1_w | to1_w] via TDM (+barrier)
  if (tid<16){
    float s=0.f;
    for (int v=0;v<8;++v) s+=wgt[tid*8+v];
    wsum[tid]=s+1e-6f;
  }
  __syncthreads();
  if (tid<128) wgt[tid]/=wsum[tid>>3];
  __syncthreads();

  const int wave=tid>>5, lane=tid&31, half=lane>>4, lrow=lane&15;
  _Float16* scr = wscr[wave];
  const _Float16* xrow = &xin[(wave*16+lrow)*96];

  // tv1 + weighted mean/var over views (intra-lane) -> cat96 in scr
  {
    float m, rs;
    row_stats96(xrow, half, m, rs);
    v16h a0=ln_frag96(xrow, 0, half, m, rs, lgA, lbA);
    v16h a1=ln_frag96(xrow,32, half, m, rs, lgA, lbA);
    v16h a2=ln_frag96(xrow,64, half, m, rs, lgA, lbA);
    const int rb = wave*16 + 8*half;
    #pragma unroll
    for (int nt=0;nt<2;++nt){
      v8f c=ZERO8;
      c=WMMA_F16(a0, load_frag(wv, nt*16+lrow, 96, 0, half), c);
      c=WMMA_F16(a1, load_frag(wv, nt*16+lrow, 96,32, half), c);
      c=WMMA_F16(a2, load_frag(wv, nt*16+lrow, 96,64, half), c);
      float mean=0.f;
      #pragma unroll
      for (int r=0;r<8;++r) mean += wgt[rb+r]*c[r];
      float var=0.f;
      #pragma unroll
      for (int r=0;r<8;++r){ float d=c[r]-mean; var+=wgt[rb+r]*d*d; }
      #pragma unroll
      for (int r=0;r<8;++r){
        scr[(r+8*half)*96 + nt*16 + lrow]      = (_Float16)c[r];
        scr[(r+8*half)*96 + 32 + nt*16 + lrow] = (_Float16)mean;
        scr[(r+8*half)*96 + 64 + nt*16 + lrow] = (_Float16)var;
      }
    }
  }
  __syncthreads();

  // to1: x = cat96 @ to1_w + b + x_input -> scr (stride 96)
  {
    v16h a0=load_frag(scr, lrow, 96, 0, half);
    v16h a1=load_frag(scr, lrow, 96,32, half);
    v16h a2=load_frag(scr, lrow, 96,64, half);
    __syncthreads();
    #pragma unroll
    for (int nt=0;nt<6;++nt){
      v8f c=ZERO8;
      c=WMMA_F16(a0, load_frag(wbuf, nt*16+lrow, 96, 0, half), c);
      c=WMMA_F16(a1, load_frag(wbuf, nt*16+lrow, 96,32, half), c);
      c=WMMA_F16(a2, load_frag(wbuf, nt*16+lrow, 96,64, half), c);
      int col=nt*16+lrow;
      float bias=bbuf[col];
      #pragma unroll
      for (int r=0;r<8;++r){
        int M=r+8*half;
        scr[M*96+col]=(_Float16)(c[r]+bias+(float)xin[(wave*16+M)*96+col]);
      }
    }
  }
  __syncthreads();

  for (int i=tid;i<256;i+=nth) bbuf[i]=n1_b1[i];
  tdm_stage(wb+OFF_N1W1, wbuf, 24576, tid);

  // n1 a: h = gelu(LN(x) @ w1 + b1) -> scr (stride 256)
  {
    float m, rs;
    row_stats96(&scr[lrow*96], half, m, rs);
    v16h a0=ln_frag96(&scr[lrow*96], 0, half, m, rs, lgB, lbB);
    v16h a1=ln_frag96(&scr[lrow*96],32, half, m, rs, lgB, lbB);
    v16h a2=ln_frag96(&scr[lrow*96],64, half, m, rs, lgB, lbB);
    __syncthreads();
    #pragma unroll
    for (int nt=0;nt<16;++nt){
      v8f c=ZERO8;
      c=WMMA_F16(a0, load_frag(wbuf, nt*16+lrow, 96, 0, half), c);
      c=WMMA_F16(a1, load_frag(wbuf, nt*16+lrow, 96,32, half), c);
      c=WMMA_F16(a2, load_frag(wbuf, nt*16+lrow, 96,64, half), c);
      int col=nt*16+lrow;
      float bias=bbuf[col];
      #pragma unroll
      for (int r=0;r<8;++r)
        scr[(r+8*half)*256+col]=(_Float16)geluf(c[r]+bias);
    }
  }
  __syncthreads();

  for (int i=tid;i<96;i+=nth) bbuf[i]=n1_b2[i];
  tdm_stage(wb+OFF_N1W2, wbuf, 24576, tid);

  // n1 b: x = h @ w2 + b2 + x_input -> global
  {
    v16h ha[8];
    #pragma unroll
    for (int kf=0;kf<8;++kf) ha[kf]=load_frag(scr, lrow, 256, kf*32, half);
    #pragma unroll
    for (int nt=0;nt<6;++nt){
      v8f c=ZERO8;
      #pragma unroll
      for (int kf=0;kf<8;++kf)
        c=WMMA_F16(ha[kf], load_frag(wbuf, nt*16+lrow, 256, kf*32, half), c);
      int col=nt*16+lrow;
      float bias=bbuf[col];
      #pragma unroll
      for (int r=0;r<8;++r){
        int M=r+8*half;
        xout[(Rbase+wave*16+M)*96+col]=c[r]+bias+(float)xin[(wave*16+M)*96+col];
      }
    }
  }
}

// ---------------- Kernel 3: block 2 + head ----------------
__global__ __launch_bounds__(256) void k3_head(
    const float* __restrict__ rgb, const float* __restrict__ fm,
    const float* __restrict__ proj_err, const float* __restrict__ pbrfeat,
    const float* __restrict__ xws,
    const float* __restrict__ gA, const float* __restrict__ bA,
    const float* __restrict__ to2_b,
    const float* __restrict__ gB, const float* __restrict__ bB,
    const float* __restrict__ n2_b1, const float* __restrict__ n2_b2,
    const float* __restrict__ gC, const float* __restrict__ bC,
    const float* __restrict__ brdf_b1, const float* __restrict__ brdf_b2,
    const _Float16* __restrict__ wb,
    float* __restrict__ out)
{
  __shared__ _Float16 xall[128*96];
  __shared__ _Float16 xin0[16*96];
  __shared__ _Float16 head[16*96];
  __shared__ _Float16 hbuf[16*256];
  __shared__ _Float16 wsh[3072 + 24576];
  __shared__ float bbuf[256];
  __shared__ float wgt[128]; __shared__ float wsum[16];
  __shared__ float lgA[96], lbA[96], lgB[96], lbB[96], lgC[96], lbC[96];

  _Float16* wv   = wsh;
  _Float16* wbuf = wsh + 3072;

  const int tid=threadIdx.x, nth=blockDim.x;
  const size_t Rbase=(size_t)blockIdx.x*128;
  const int pixbase=blockIdx.x*16;

  for (int i=tid;i<128*96;i+=nth) xall[i]=(_Float16)xws[Rbase*96+i];
  for (int i=tid;i<16*96;i+=nth){
    int row=i/96, col=i%96;
    int pix=pixbase+row;
    size_t R0=(size_t)pix*8;
    float v;
    if (col<3)       v=rgb[R0*3+col];
    else if (col<64) v=fm[(size_t)pix*61+(col-3)];
    else             v=pbrfeat[R0*32+(col-64)];
    xin0[i]=(_Float16)v;
  }
  if (tid<128){
    float pe=proj_err[Rbase+tid];
    wgt[tid]=-fminf(log10f(fabsf(pe)+1e-6f),0.f);
  }
  for (int i=tid;i<96;i+=nth){
    bbuf[i]=to2_b[i];
    lgA[i]=gA[i]; lbA[i]=bA[i];
    lgB[i]=gB[i]; lbB[i]=bB[i];
    lgC[i]=gC[i]; lbC[i]=bC[i];
  }
  tdm_stage(wb+OFF_TV2, wsh, SZ_TV2TO2, tid);   // [tv2_w | to2_w] via TDM (+barrier)
  if (tid<16){ float s=0.f; for(int v=0;v<8;++v) s+=wgt[tid*8+v]; wsum[tid]=s+1e-6f; }
  __syncthreads();
  if (tid<128) wgt[tid]/=wsum[tid>>3];
  __syncthreads();

  const int wave=tid>>5, lane=tid&31, half=lane>>4, lrow=lane&15;

  // tv2 + moments -> head rows [v_view0 | mean | var] (one row per pixel)
  {
    const _Float16* xr=&xall[(wave*16+lrow)*96];
    float m, rs;
    row_stats96(xr, half, m, rs);
    v16h a0=ln_frag96(xr, 0, half, m, rs, lgA, lbA);
    v16h a1=ln_frag96(xr,32, half, m, rs, lgA, lbA);
    v16h a2=ln_frag96(xr,64, half, m, rs, lgA, lbA);
    const int rb=wave*16+8*half;
    const int prow=2*wave+half;
    #pragma unroll
    for (int nt=0;nt<2;++nt){
      v8f c=ZERO8;
      c=WMMA_F16(a0, load_frag(wv, nt*16+lrow, 96, 0, half), c);
      c=WMMA_F16(a1, load_frag(wv, nt*16+lrow, 96,32, half), c);
      c=WMMA_F16(a2, load_frag(wv, nt*16+lrow, 96,64, half), c);
      float mean=0.f;
      #pragma unroll
      for (int r=0;r<8;++r) mean+=wgt[rb+r]*c[r];
      float var=0.f;
      #pragma unroll
      for (int r=0;r<8;++r){ float d=c[r]-mean; var+=wgt[rb+r]*d*d; }
      head[prow*96 + nt*16+lrow]      = (_Float16)c[0];   // c[0] = view-0 row
      head[prow*96 + 32 + nt*16+lrow] = (_Float16)mean;
      head[prow*96 + 64 + nt*16+lrow] = (_Float16)var;
    }
  }
  __syncthreads();

  // to2 : x0 = head @ to2_w + b + xin0 -> head (n-tiles over waves 0..5)
  {
    v16h a0=load_frag(head, lrow, 96, 0, half);
    v16h a1=load_frag(head, lrow, 96,32, half);
    v16h a2=load_frag(head, lrow, 96,64, half);
    __syncthreads();
    if (wave<6){
      int nt=wave;
      v8f c=ZERO8;
      c=WMMA_F16(a0, load_frag(wbuf, nt*16+lrow, 96, 0, half), c);
      c=WMMA_F16(a1, load_frag(wbuf, nt*16+lrow, 96,32, half), c);
      c=WMMA_F16(a2, load_frag(wbuf, nt*16+lrow, 96,64, half), c);
      int col=nt*16+lrow;
      float bias=bbuf[col];
      #pragma unroll
      for (int r=0;r<8;++r){
        int M=r+8*half;
        head[M*96+col]=(_Float16)(c[r]+bias+(float)xin0[M*96+col]);
      }
    }
  }
  __syncthreads();

  for (int i=tid;i<256;i+=nth) bbuf[i]=n2_b1[i];
  tdm_stage(wb+OFF_N2W1, wbuf, 24576, tid);

  // n2 a: hbuf = gelu(LN(x0) @ w1 + b1)  (2 n-tiles per wave)
  {
    float m, rs;
    row_stats96(&head[lrow*96], half, m, rs);
    v16h a0=ln_frag96(&head[lrow*96], 0, half, m, rs, lgB, lbB);
    v16h a1=ln_frag96(&head[lrow*96],32, half, m, rs, lgB, lbB);
    v16h a2=ln_frag96(&head[lrow*96],64, half, m, rs, lgB, lbB);
    #pragma unroll
    for (int t=0;t<2;++t){
      int nt=wave*2+t;
      v8f c=ZERO8;
      c=WMMA_F16(a0, load_frag(wbuf, nt*16+lrow, 96, 0, half), c);
      c=WMMA_F16(a1, load_frag(wbuf, nt*16+lrow, 96,32, half), c);
      c=WMMA_F16(a2, load_frag(wbuf, nt*16+lrow, 96,64, half), c);
      int col=nt*16+lrow;
      float bias=bbuf[col];
      #pragma unroll
      for (int r=0;r<8;++r)
        hbuf[(r+8*half)*256+col]=(_Float16)geluf(c[r]+bias);
    }
  }
  __syncthreads();

  for (int i=tid;i<96;i+=nth) bbuf[i]=n2_b2[i];
  tdm_stage(wb+OFF_N2W2, wbuf, 24576, tid);

  // n2 b: x2 = hbuf @ w2 + b2 + xin0 -> head
  {
    v16h ha[8];
    #pragma unroll
    for (int kf=0;kf<8;++kf) ha[kf]=load_frag(hbuf, lrow, 256, kf*32, half);
    if (wave<6){
      int nt=wave;
      v8f c=ZERO8;
      #pragma unroll
      for (int kf=0;kf<8;++kf)
        c=WMMA_F16(ha[kf], load_frag(wbuf, nt*16+lrow, 256, kf*32, half), c);
      int col=nt*16+lrow;
      float bias=bbuf[col];
      #pragma unroll
      for (int r=0;r<8;++r){
        int M=r+8*half;
        head[M*96+col]=(_Float16)(c[r]+bias+(float)xin0[M*96+col]);
      }
    }
  }
  __syncthreads();

  for (int i=tid;i<128;i+=nth) bbuf[i]=brdf_b1[i];
  tdm_stage(wb+OFF_BW1, wbuf, 12288, tid);

  // brdf a: hbuf(stride 128) = gelu(LN(x2) @ w1 + b1)  (1 n-tile per wave)
  {
    float m, rs;
    row_stats96(&head[lrow*96], half, m, rs);
    v16h a0=ln_frag96(&head[lrow*96], 0, half, m, rs, lgC, lbC);
    v16h a1=ln_frag96(&head[lrow*96],32, half, m, rs, lgC, lbC);
    v16h a2=ln_frag96(&head[lrow*96],64, half, m, rs, lgC, lbC);
    int nt=wave;
    v8f c=ZERO8;
    c=WMMA_F16(a0, load_frag(wbuf, nt*16+lrow, 96, 0, half), c);
    c=WMMA_F16(a1, load_frag(wbuf, nt*16+lrow, 96,32, half), c);
    c=WMMA_F16(a2, load_frag(wbuf, nt*16+lrow, 96,64, half), c);
    int col=nt*16+lrow;
    float bias=bbuf[col];
    #pragma unroll
    for (int r=0;r<8;++r)
      hbuf[(r+8*half)*128+col]=(_Float16)geluf(c[r]+bias);
  }
  __syncthreads();

  for (int i=tid;i<128;i+=nth) bbuf[i]=brdf_b2[i];
  tdm_stage(wb+OFF_BW2, wbuf, 16384, tid);

  // brdf b: out = hbuf @ w2 + b2
  {
    v16h ha[4];
    #pragma unroll
    for (int kf=0;kf<4;++kf) ha[kf]=load_frag(hbuf, lrow, 128, kf*32, half);
    int nt=wave;
    v8f c=ZERO8;
    #pragma unroll
    for (int kf=0;kf<4;++kf)
      c=WMMA_F16(ha[kf], load_frag(wbuf, nt*16+lrow, 128, kf*32, half), c);
    int col=nt*16+lrow;
    float bias=bbuf[col];
    #pragma unroll
    for (int r=0;r<8;++r){
      int M=r+8*half;
      out[((size_t)pixbase+M)*128+col]=c[r]+bias;
    }
  }
}

extern "C" void kernel_launch(void* const* d_in, const int* in_sizes, int n_in,
                              void* d_out, int out_size, void* d_ws, size_t ws_size,
                              hipStream_t stream)
{
  const float* rgb   =(const float*)d_in[0];
  const float* fm    =(const float*)d_in[1];
  const float* vdir  =(const float*)d_in[2];
  const float* perr  =(const float*)d_in[3];
  const float* nrm   =(const float*)d_in[4];
  const float* DL    =(const float*)d_in[5];
  const float* p_ln_g=(const float*)d_in[6];
  const float* p_ln_b=(const float*)d_in[7];
  const float* p_w1  =(const float*)d_in[8];
  const float* p_b1  =(const float*)d_in[9];
  const float* p_w2  =(const float*)d_in[10];
  const float* p_b2  =(const float*)d_in[11];
  const float* p_w3  =(const float*)d_in[12];
  const float* p_b3  =(const float*)d_in[13];
  const float* p_w4  =(const float*)d_in[14];
  const float* p_b4  =(const float*)d_in[15];
  const float* tv1g  =(const float*)d_in[16];
  const float* tv1b  =(const float*)d_in[17];
  const float* tv1w  =(const float*)d_in[18];
  const float* to1w  =(const float*)d_in[19];
  const float* to1b  =(const float*)d_in[20];
  const float* n1g   =(const float*)d_in[21];
  const float* n1lb  =(const float*)d_in[22];
  const float* n1w1  =(const float*)d_in[23];
  const float* n1b1  =(const float*)d_in[24];
  const float* n1w2  =(const float*)d_in[25];
  const float* n1b2  =(const float*)d_in[26];
  const float* tv2g  =(const float*)d_in[27];
  const float* tv2b  =(const float*)d_in[28];
  const float* tv2w  =(const float*)d_in[29];
  const float* to2w  =(const float*)d_in[30];
  const float* to2b  =(const float*)d_in[31];
  const float* n2g   =(const float*)d_in[32];
  const float* n2lb  =(const float*)d_in[33];
  const float* n2w1  =(const float*)d_in[34];
  const float* n2b1  =(const float*)d_in[35];
  const float* n2w2  =(const float*)d_in[36];
  const float* n2b2  =(const float*)d_in[37];
  const float* bdg   =(const float*)d_in[38];
  const float* bdb   =(const float*)d_in[39];
  const float* bdw1  =(const float*)d_in[40];
  const float* bdb1  =(const float*)d_in[41];
  const float* bdw2  =(const float*)d_in[42];
  const float* bdb2  =(const float*)d_in[43];
  float* out=(float*)d_out;

  float*     pbrfeat=(float*)d_ws;                               // 16 MB f32
  float*     xws    = pbrfeat + (size_t)16384*8*32;              // 48 MB f32
  _Float16*  wblob  = (_Float16*)((char*)d_ws + (64ull<<20));    // 320 KB f16

  k0_prep<<<256,256,0,stream>>>(p_w1,p_w2,p_w3,p_w4,
                                tv1w,to1w,n1w1,n1w2,
                                tv2w,to2w,n2w1,n2w2,
                                bdw1,bdw2, wblob);
  k1_pbr<<<16384,256,0,stream>>>(vdir,nrm,DL,p_ln_g,p_ln_b,
                                 p_b1,p_b2,p_b3,p_b4, wblob, pbrfeat);
  k2_block1<<<1024,256,0,stream>>>(rgb,fm,perr,pbrfeat,
                                   tv1g,tv1b,to1b,
                                   n1g,n1lb,n1b1,n1b2, wblob, xws);
  k3_head<<<1024,256,0,stream>>>(rgb,fm,perr,pbrfeat,xws,
                                 tv2g,tv2b,to2b,
                                 n2g,n2lb,n2b1,n2b2,
                                 bdg,bdb,bdb1,bdb2, wblob, out);
}
